// QLoRALinear_35527969472751
// MI455X (gfx1250) — compile-verified, matrix operand order checked by reading, not run on hardware
//
#include <hip/hip_runtime.h>

// ---------------------------------------------------------------------------
// QLoRA linear for MI455X (gfx1250, wave32, WMMA, async LDS copies).
//   out[M,N] = x[M,K] @ Weff[N,K]^T + bias,  Weff = NF4deq(W) + 4 * B @ A
// Pass 0: x fp32 -> bf16 (once; keeps conversions out of the GEMM hot loop).
// Pass 1: fuse NF4 dequant + rank-8 LoRA into bf16 Weff.
// Pass 2: block-tiled bf16 WMMA GEMM; tiles staged with
//         global_load_async_to_lds_b128 (ASYNCcnt), fp32 accumulate.
// Workspace layout: [0, 33.5MB) Weff bf16, [33.5MB, 100.7MB) x bf16.
// ---------------------------------------------------------------------------

typedef __attribute__((ext_vector_type(16))) __bf16 v16bf;
typedef __attribute__((ext_vector_type(2)))  __bf16 v2bf;
typedef __attribute__((ext_vector_type(8)))  float  v8f;

constexpr int   IN      = 4096;
constexpr int   OUT     = 4096;
constexpr int   M_TOTAL = 4 * 2048;          // B * S
constexpr int   R       = 8;
constexpr float SCALING = 32.0f / 8.0f;      // alpha / r

constexpr int BM = 128, BN = 128, BK = 32;   // block tile / K step (WMMA-native)
constexpr int LDS_STRIDE = 40;               // 32 + 8 pad bf16 = 80B rows:
                                             // 20*r mod 64 distinct for r=0..15
                                             // -> conflict-free ds_load_b128,
                                             // rows stay 16B-aligned.

__constant__ float NF4[16] = {
    -1.0f, -0.6961928009986877f, -0.5250730514526367f, -0.39491748809814453f,
    -0.28444138169288635f, -0.18477343022823334f, -0.09105003625154495f, 0.0f,
    0.07958029955625534f, 0.16093020141124725f, 0.24611230194568634f,
    0.33791524171829224f, 0.44070982933044434f, 0.5626170039176941f,
    0.7229568362236023f, 1.0f};

// Packed fp32x2 -> bf16x2 (targets v_cvt_pk_bf16_f32).
__device__ __forceinline__ unsigned pack2_bf16(float lo, float hi) {
  v2bf p = {(__bf16)lo, (__bf16)hi};
  return __builtin_bit_cast(unsigned, p);
}

// ---------------------------------------------------------------------------
// Pass 0: x fp32 -> bf16, 8 elements per thread.
// ---------------------------------------------------------------------------
__global__ __launch_bounds__(256)
void f32_to_bf16_stream(const float* __restrict__ in,
                        unsigned short* __restrict__ outb) {
  const int base = (blockIdx.x * 256 + threadIdx.x) * 8;
  const float4 a = *reinterpret_cast<const float4*>(in + base);
  const float4 b = *reinterpret_cast<const float4*>(in + base + 4);
  uint4 pk;
  pk.x = pack2_bf16(a.x, a.y);
  pk.y = pack2_bf16(a.z, a.w);
  pk.z = pack2_bf16(b.x, b.y);
  pk.w = pack2_bf16(b.z, b.w);
  *reinterpret_cast<uint4*>(outb + base) = pk;
}

// ---------------------------------------------------------------------------
// Pass 1: Weff[o][i] = NF4[q] * scale(group) + 4 * sum_r B[o][r]*A[r][i]
// 8 elements per thread. lora_A/B (128 KB each) live entirely in L2.
// ---------------------------------------------------------------------------
__global__ __launch_bounds__(256)
void nf4_dequant_fuse_lora(const int* __restrict__ qw,
                           const float* __restrict__ qscale,
                           const float* __restrict__ loraA,
                           const float* __restrict__ loraB,
                           unsigned short* __restrict__ weff) {
  const int t    = blockIdx.x * 256 + threadIdx.x;
  const int base = t * 8;                    // flat index into [OUT, IN]
  const int o    = base >> 12;               // / IN
  const int i    = base & (IN - 1);
  const float scale = qscale[base >> 6];     // group of 64, 8 | 64 -> uniform

  const int4 q0 = *reinterpret_cast<const int4*>(qw + base);
  const int4 q1 = *reinterpret_cast<const int4*>(qw + base + 4);
  float w[8];
  w[0] = NF4[q0.x & 15]; w[1] = NF4[q0.y & 15];
  w[2] = NF4[q0.z & 15]; w[3] = NF4[q0.w & 15];
  w[4] = NF4[q1.x & 15]; w[5] = NF4[q1.y & 15];
  w[6] = NF4[q1.z & 15]; w[7] = NF4[q1.w & 15];

  float l[8] = {0.f, 0.f, 0.f, 0.f, 0.f, 0.f, 0.f, 0.f};
#pragma unroll
  for (int r = 0; r < R; ++r) {
    const float br = loraB[o * R + r];
    const float4 a0 = *reinterpret_cast<const float4*>(loraA + r * IN + i);
    const float4 a1 = *reinterpret_cast<const float4*>(loraA + r * IN + i + 4);
    l[0] = fmaf(br, a0.x, l[0]); l[1] = fmaf(br, a0.y, l[1]);
    l[2] = fmaf(br, a0.z, l[2]); l[3] = fmaf(br, a0.w, l[3]);
    l[4] = fmaf(br, a1.x, l[4]); l[5] = fmaf(br, a1.y, l[5]);
    l[6] = fmaf(br, a1.z, l[6]); l[7] = fmaf(br, a1.w, l[7]);
  }

  float v[8];
#pragma unroll
  for (int j = 0; j < 8; ++j) v[j] = fmaf(w[j], scale, SCALING * l[j]);

  uint4 pk;
  pk.x = pack2_bf16(v[0], v[1]);
  pk.y = pack2_bf16(v[2], v[3]);
  pk.z = pack2_bf16(v[4], v[5]);
  pk.w = pack2_bf16(v[6], v[7]);
  *reinterpret_cast<uint4*>(weff + base) = pk;
}

// ---------------------------------------------------------------------------
// Fragment load from LDS, matching ISA 16-bit A/B VGPR layout (16x32 MxK):
//   lanes 0-15 : row = lane,    K in {0..7}  U {16..23}
//   lanes 16-31: row = lane-16, K in {8..15} U {24..31}
// Two 16B ds loads per fragment, bank-conflict-free with LDS_STRIDE=40.
// ---------------------------------------------------------------------------
__device__ __forceinline__ v16bf frag_ld(const unsigned short* tileBase,
                                         int r16, int half) {
  const unsigned short* p = tileBase + r16 * LDS_STRIDE + half * 8;
  union { uint4 u[2]; v16bf v; } f;
  f.u[0] = *reinterpret_cast<const uint4*>(p);        // K chunk +0
  f.u[1] = *reinterpret_cast<const uint4*>(p + 16);   // K chunk +16
  return f.v;
}

// ---------------------------------------------------------------------------
// Pass 2: C[128x128] per block, 8 waves, wave tile 64x32 = 4x2 WMMA tiles.
// Tiles staged via async copy (ASYNCcnt) into double-buffered LDS:
// per thread, one 16-bf16 strip of A and of B per K step = 2x b128 async each
// (INST_OFFSET is added to both LDS and global addresses, so strip chunk 1 is
// just "offset:16"). s_wait_asynccnt 0 fences the copies before the barrier.
// ---------------------------------------------------------------------------
__global__ __launch_bounds__(256)
void qlora_wmma_gemm(const unsigned short* __restrict__ xbf,
                     const unsigned short* __restrict__ weff,
                     const float* __restrict__ bias,
                     float* __restrict__ out) {
  __shared__ unsigned short ldsA[2][BM * LDS_STRIDE];   // 2 x 10 KB
  __shared__ unsigned short ldsB[2][BN * LDS_STRIDE];   // 2 x 10 KB

  const int tid   = threadIdx.x;
  const int lane  = tid & 31;
  const int wid   = tid >> 5;
  const int r16   = lane & 15;
  const int half  = lane >> 4;
  const int wMoff = (wid >> 2) * 64;                    // wave grid 2(M) x 4(N)
  const int wNoff = (wid & 3) * 32;

  const int blockN = blockIdx.x * BN;
  const int blockM = blockIdx.y * BM;

  // staging map: thread -> (row, 16-wide K strip) of the 128x32 tile
  const int sRow = tid >> 1;
  const int sCol = (tid & 1) * 16;

  const unsigned short* gA = xbf  + (size_t)(blockM + sRow) * IN + sCol;
  const unsigned short* gB = weff + (size_t)(blockN + sRow) * IN + sCol;

  // LDS byte addresses for this thread's strips (generic ptr low 32 bits ==
  // LDS offset per the aperture mapping; rows are 16B-aligned).
  unsigned aoff[2], boff[2];
#pragma unroll
  for (int b = 0; b < 2; ++b) {
    aoff[b] = (unsigned)(uintptr_t)&ldsA[b][sRow * LDS_STRIDE + sCol];
    boff[b] = (unsigned)(uintptr_t)&ldsB[b][sRow * LDS_STRIDE + sCol];
  }

  // async copy of one K tile into LDS buffer `buf`
  auto stage = [&](int buf, int kb) {
    const unsigned short* pa = gA + kb;
    const unsigned short* pb = gB + kb;
    asm volatile("global_load_async_to_lds_b128 %0, %1, off"
                 :: "v"(aoff[buf]), "v"(pa) : "memory");
    asm volatile("global_load_async_to_lds_b128 %0, %1, off offset:16"
                 :: "v"(aoff[buf]), "v"(pa) : "memory");
    asm volatile("global_load_async_to_lds_b128 %0, %1, off"
                 :: "v"(boff[buf]), "v"(pb) : "memory");
    asm volatile("global_load_async_to_lds_b128 %0, %1, off offset:16"
                 :: "v"(boff[buf]), "v"(pb) : "memory");
  };

  const v8f vzero = {0.f, 0.f, 0.f, 0.f, 0.f, 0.f, 0.f, 0.f};
  v8f acc[4][2];
#pragma unroll
  for (int mi = 0; mi < 4; ++mi)
#pragma unroll
    for (int ni = 0; ni < 2; ++ni) acc[mi][ni] = vzero;

  // ---- prologue: stage K-tile 0 into buffer 0 ----
  stage(0, 0);
  asm volatile("s_wait_asynccnt 0x0" ::: "memory");
  __syncthreads();

  constexpr int KT = IN / BK;                           // 128 steps
  for (int kt = 0; kt < KT; ++kt) {
    const int  cur      = kt & 1;
    const int  nxt      = cur ^ 1;
    const bool haveNext = (kt + 1) < KT;

    if (haveNext) {
      stage(nxt, (kt + 1) * BK);                        // overlap with WMMAs
      if (kt + 2 < KT)                                  // global_prefetch_b8
        __builtin_prefetch(gB + (kt + 2) * BK, 0, 1);
    }

    // ---- compute: 6 fragment loads + 8 WMMAs from LDS[cur] ----
    v16bf af[4], bf[2];
#pragma unroll
    for (int mi = 0; mi < 4; ++mi)
      af[mi] = frag_ld(&ldsA[cur][(wMoff + mi * 16) * LDS_STRIDE], r16, half);
#pragma unroll
    for (int ni = 0; ni < 2; ++ni)
      bf[ni] = frag_ld(&ldsB[cur][(wNoff + ni * 16) * LDS_STRIDE], r16, half);

#pragma unroll
    for (int mi = 0; mi < 4; ++mi)
#pragma unroll
      for (int ni = 0; ni < 2; ++ni)
        acc[mi][ni] = __builtin_amdgcn_wmma_f32_16x16x32_bf16(
            /*neg_a=*/false, af[mi], /*neg_b=*/false, bf[ni],
            /*c_mod=*/(short)0, acc[mi][ni],
            /*reuse_a=*/false, /*reuse_b=*/false);

    asm volatile("s_wait_asynccnt 0x0" ::: "memory");   // copies into nxt done
    __syncthreads();
  }

  // ---- epilogue: C/D layout is VGPR r -> M = r (lanes 0-15) / 8+r (16-31),
  //      N = lane & 15. Fuse bias, store fp32 rows (coalesced 64B/16 lanes).
#pragma unroll
  for (int ni = 0; ni < 2; ++ni) {
    const int   gn = blockN + wNoff + ni * 16 + r16;
    const float bv = bias[gn];
#pragma unroll
    for (int mi = 0; mi < 4; ++mi) {
      const int gm = blockM + wMoff + mi * 16 + half * 8;
      float* op = out + (size_t)gm * OUT + gn;
#pragma unroll
      for (int r = 0; r < 8; ++r)
        op[(size_t)r * OUT] = acc[mi][ni][r] + bv;
    }
  }
}

// ---------------------------------------------------------------------------
// Launch: pass 0 (x->bf16) + pass 1 (dequant+LoRA fold) -> pass 2 (WMMA GEMM).
// Same stream, graph-capture safe (no sync, no alloc; scratch = d_ws).
// ---------------------------------------------------------------------------
extern "C" void kernel_launch(void* const* d_in, const int* in_sizes, int n_in,
                              void* d_out, int out_size, void* d_ws, size_t ws_size,
                              hipStream_t stream) {
  (void)in_sizes; (void)n_in; (void)out_size; (void)ws_size;
  const float* x       = (const float*)d_in[0];
  const int*   qweight = (const int*)d_in[1];
  const float* qscale  = (const float*)d_in[2];
  const float* bias    = (const float*)d_in[3];
  const float* loraA   = (const float*)d_in[4];
  const float* loraB   = (const float*)d_in[5];
  float*       out     = (float*)d_out;

  unsigned short* weff = (unsigned short*)d_ws;              // OUT*IN bf16
  unsigned short* xbf  = weff + (size_t)OUT * IN;            // M*IN bf16

  // Pass 0: M*IN/8 threads
  f32_to_bf16_stream<<<(M_TOTAL * (IN / 8)) / 256, 256, 0, stream>>>(x, xbf);
  // Pass 1: OUT*IN/8 threads
  nf4_dequant_fuse_lora<<<(OUT * (IN / 8)) / 256, 256, 0, stream>>>(
      qweight, qscale, loraA, loraB, weff);
  // Pass 2: grid (N/128, M/128)
  dim3 grid(OUT / BN, M_TOTAL / BM);
  qlora_wmma_gemm<<<grid, 256, 0, stream>>>(xbf, weff, bias, out);
}
